// SelfTransformerBlcok_37142877176426
// MI455X (gfx1250) — matrix operand order repaired and checked by reading, not compiled, not run on hardware
//
#include <hip/hip_runtime.h>
#include <hip/hip_bf16.h>
#include <math.h>

// ---------------------------------------------------------------------------
// Swin-style block, fused for gfx1250 (wave32, WMMA 16x16x32 f16 -> f32 acc)
// ---------------------------------------------------------------------------

typedef __attribute__((ext_vector_type(16))) _Float16 v16h;
typedef __attribute__((ext_vector_type(8)))  _Float16 v8h;
typedef __attribute__((ext_vector_type(8)))  float    v8f;

#define DIM   192
#define HID   768
#define IMG_H 256
#define IMG_W 256
#define WH    8
#define SH    4
#define WINL  64   // tokens per window

__device__ __forceinline__ v8f wmma32(v16h a, v16h b, v8f c) {
  // D = A(16x32 f16) x B(32x16 f16) + C(16x16 f32)
  return __builtin_amdgcn_wmma_f32_16x16x32_f16(false, a, false, b,
                                                (short)0, c, false, false);
}

__device__ __forceinline__ v16h cat16(v8h lo, v8h hi) {
  return __builtin_shufflevector(lo, hi, 0, 1, 2, 3, 4, 5, 6, 7,
                                 8, 9, 10, 11, 12, 13, 14, 15);
}

// A-fragment: 16x32 tile, row-major source, row stride `stride` (halfs).
// Per ISA 7.12.2: lane m=lane%16 holds row m; VGPR r<4 -> K = half*8 + 2r,
// VGPR r>=4 -> K = 16 + half*8 + 2(r-4).  Two contiguous 8-half chunks.
__device__ __forceinline__ v16h load_a_frag(const _Float16* src, int stride,
                                            int row0, int k0, int lane) {
  int m = lane & 15, half = lane >> 4;
  const _Float16* p = src + (size_t)(row0 + m) * stride + k0 + half * 8;
  v8h lo = *(const v8h*)p;         // K = half*8 + 0..7
  v8h hi = *(const v8h*)(p + 16);  // K = 16 + half*8 + 0..7
  return cat16(lo, hi);
}

// B-fragment: 32x16 tile where the source S is stored transposed,
// S[n][k] == B[k][n], row stride `stride` (halfs). Lane n=lane%16 = column,
// elements 0..7 hold K = half*8 + 0..7, elements 8..15 hold K = 16 + half*8 + 0..7.
// -> two contiguous 8-half (16B) loads per lane (same K order as A).
__device__ __forceinline__ v16h load_bt_frag(const _Float16* src, int stride,
                                             int n0, int k0, int lane) {
  int n = lane & 15, half = lane >> 4;
  const _Float16* p = src + (size_t)(n0 + n) * stride + k0 + half * 8;
  return cat16(*(const v8h*)p, *(const v8h*)(p + 16));
}

__device__ __forceinline__ int region8(int p) {
  // h_slices boundaries: [0,248) [248,252) [252,256)
  return (p < IMG_H - WH) ? 0 : ((p < IMG_H - SH) ? 1 : 2);
}

// ---------------------------------------------------------------------------
// Weight f32 -> f16 transpose:  dst[n*K + k] = src[k*N + n]
// ---------------------------------------------------------------------------
__global__ void transpose_cvt(const float* __restrict__ src,
                              _Float16* __restrict__ dst, int K, int N) {
  int idx = blockIdx.x * blockDim.x + threadIdx.x;
  if (idx >= K * N) return;
  int n = idx / K, k = idx - n * K;
  dst[idx] = (_Float16)src[(size_t)k * N + n];
}

// ---------------------------------------------------------------------------
// Kernel 1: LN1 + shifted-window attention + proj + residual
// one workgroup per window; 8 waves; 120 KB dynamic LDS
// ---------------------------------------------------------------------------
__global__ __launch_bounds__(256) void attn_kernel(
    const float* __restrict__ fmap,
    const _Float16* __restrict__ wqT, const _Float16* __restrict__ wkT,
    const _Float16* __restrict__ wvT, const _Float16* __restrict__ wpT,
    const float* __restrict__ bq, const float* __restrict__ bk,
    const float* __restrict__ bv, const float* __restrict__ bp,
    const float* __restrict__ g1, const float* __restrict__ be1,
    float* __restrict__ out) {
  extern __shared__ __align__(16) char smem[];
  _Float16* xln = (_Float16*)smem;           // 64x192 f16 (reused as attn-out)
  _Float16* qsm = xln + 64 * DIM;            // 64x192 f16 (q, pre-scaled)
  _Float16* ksm = qsm + 64 * DIM;            // 64x192 f16 (k)
  _Float16* vtm = ksm + 64 * DIM;            // 192x64 f16 (v transposed)
  float*    scm = (float*)(smem + 98304);    // 64x64 f32 scores
  _Float16* pmm = (_Float16*)(smem + 114688);// 64x64 f16 probs

  const int blk  = blockIdx.x;
  const int bimg = blk >> 10;          // image index (0..1)
  const int wy   = (blk >> 5) & 31;    // window row
  const int wx   = blk & 31;           // window col
  const int tid  = threadIdx.x;
  const int lane = tid & 31;
  const int wave = tid >> 5;
  const int n16  = lane & 15;
  const int hb   = (lane >> 4) * 8;

  // ---- Phase 0: load (with roll -4,-4), LayerNorm1, cvt f16 -> LDS --------
  {
    int tok = tid >> 2;               // 0..63
    int q4  = tid & 3;                // 4 lanes cooperate per token
    int wi = tok >> 3, wj = tok & 7;
    int gh = (wy * WH + wi + SH) & (IMG_H - 1);
    int gw = (wx * WH + wj + SH) & (IMG_W - 1);
    const float* src =
        fmap + (((size_t)bimg * IMG_H + gh) * IMG_W + gw) * DIM + q4 * 48;
    float4 v[12];
    float s = 0.f, ss = 0.f;
#pragma unroll
    for (int r = 0; r < 12; ++r) {
      v[r] = ((const float4*)src)[r];
      s  += v[r].x + v[r].y + v[r].z + v[r].w;
      ss += v[r].x * v[r].x + v[r].y * v[r].y + v[r].z * v[r].z +
            v[r].w * v[r].w;
    }
#pragma unroll
    for (int off = 1; off < 4; off <<= 1) {
      s  += __shfl_xor(s, off, 32);
      ss += __shfl_xor(ss, off, 32);
    }
    float mu  = s * (1.f / DIM);
    float inv = rsqrtf(ss * (1.f / DIM) - mu * mu + 1e-5f);
    _Float16* dst = xln + tok * DIM + q4 * 48;
#pragma unroll
    for (int r = 0; r < 12; ++r) {
      int c = q4 * 48 + r * 4;
      dst[r * 4 + 0] = (_Float16)((v[r].x - mu) * inv * g1[c + 0] + be1[c + 0]);
      dst[r * 4 + 1] = (_Float16)((v[r].y - mu) * inv * g1[c + 1] + be1[c + 1]);
      dst[r * 4 + 2] = (_Float16)((v[r].z - mu) * inv * g1[c + 2] + be1[c + 2]);
      dst[r * 4 + 3] = (_Float16)((v[r].w - mu) * inv * g1[c + 3] + be1[c + 3]);
    }
  }
  __syncthreads();

  // ---- Phase 1: Q,K,V = X @ W^T + b  (36 column-tile tasks) ---------------
  const float qscale = 0.07216878364870323f;  // 192^-0.5
  for (int task = wave; task < 36; task += 8) {
    int mat = task / 12;         // 0=q 1=k 2=v
    int nt  = task - mat * 12;
    const _Float16* wT = (mat == 0) ? wqT : (mat == 1) ? wkT : wvT;
    const float* bias  = (mat == 0) ? bq : (mat == 1) ? bk : bv;
    v8f acc[4] = {};
#pragma unroll
    for (int ks = 0; ks < 6; ++ks) {
      v16h b = load_bt_frag(wT, DIM, nt * 16, ks * 32, lane);
#pragma unroll
      for (int m = 0; m < 4; ++m)
        acc[m] = wmma32(load_a_frag(xln, DIM, m * 16, ks * 32, lane), b, acc[m]);
    }
    float bval = bias[nt * 16 + n16];
#pragma unroll
    for (int m = 0; m < 4; ++m)
#pragma unroll
      for (int r = 0; r < 8; ++r) {
        int row = m * 16 + hb + r, col = nt * 16 + n16;
        float val = acc[m][r] + bval;
        if (mat == 0)      qsm[row * DIM + col] = (_Float16)(val * qscale);
        else if (mat == 1) ksm[row * DIM + col] = (_Float16)val;
        else               vtm[col * WINL + row] = (_Float16)val;  // transposed
      }
  }
  __syncthreads();

  // ---- Phase 2: S = q @ k^T  (16 tiles of 16x16) --------------------------
  for (int task = wave; task < 16; task += 8) {
    int mt = task >> 2, nt = task & 3;
    v8f acc = {};
#pragma unroll
    for (int ks = 0; ks < 6; ++ks) {
      v16h a = load_a_frag(qsm, DIM, mt * 16, ks * 32, lane);
      v16h b = load_bt_frag(ksm, DIM, nt * 16, ks * 32, lane);  // B = k^T
      acc = wmma32(a, b, acc);
    }
#pragma unroll
    for (int r = 0; r < 8; ++r)
      scm[(mt * 16 + hb + r) * WINL + nt * 16 + n16] = acc[r];
  }
  __syncthreads();

  // ---- Phase 3: shifted-window mask + softmax (one thread per row) --------
  if (tid < WINL) {
    int row = tid;
    int idr = region8(wy * WH + (row >> 3)) * 3 + region8(wx * WH + (row & 7));
    float mx = -1e30f;
    for (int j = 0; j < WINL; ++j) {
      int idj = region8(wy * WH + (j >> 3)) * 3 + region8(wx * WH + (j & 7));
      float v = scm[row * WINL + j] + ((idj == idr) ? 0.f : -100.f);
      mx = fmaxf(mx, v);
    }
    float sum = 0.f;
    for (int j = 0; j < WINL; ++j) {
      int idj = region8(wy * WH + (j >> 3)) * 3 + region8(wx * WH + (j & 7));
      float v = scm[row * WINL + j] + ((idj == idr) ? 0.f : -100.f);
      float e = __expf(v - mx);
      scm[row * WINL + j] = e;
      sum += e;
    }
    float inv = 1.f / sum;
    for (int j = 0; j < WINL; ++j)
      pmm[row * WINL + j] = (_Float16)(scm[row * WINL + j] * inv);
  }
  __syncthreads();

  // ---- Phase 4: O = P @ V  (48 tiles, K=64) -> reuse xln as attn-out ------
  for (int task = wave; task < 48; task += 8) {
    int mt = task / 12, nt = task - mt * 12;
    v8f acc = {};
#pragma unroll
    for (int ks = 0; ks < 2; ++ks) {
      v16h a = load_a_frag(pmm, WINL, mt * 16, ks * 32, lane);
      v16h b = load_bt_frag(vtm, WINL, nt * 16, ks * 32, lane);  // v stored ^T
      acc = wmma32(a, b, acc);
    }
#pragma unroll
    for (int r = 0; r < 8; ++r)
      xln[(mt * 16 + hb + r) * DIM + nt * 16 + n16] = (_Float16)acc[r];
  }
  __syncthreads();

  // ---- Phase 5: Y = O @ Wp^T + bp, un-roll + residual -> d_out ------------
  for (int task = wave; task < 48; task += 8) {
    int mt = task / 12, nt = task - mt * 12;
    v8f acc = {};
#pragma unroll
    for (int ks = 0; ks < 6; ++ks) {
      v16h a = load_a_frag(xln, DIM, mt * 16, ks * 32, lane);
      v16h b = load_bt_frag(wpT, DIM, nt * 16, ks * 32, lane);
      acc = wmma32(a, b, acc);
    }
    int col = nt * 16 + n16;
    float bval = bp[col];
#pragma unroll
    for (int r = 0; r < 8; ++r) {
      int tok = mt * 16 + hb + r;
      int gh = (wy * WH + (tok >> 3) + SH) & (IMG_H - 1);
      int gw = (wx * WH + (tok & 7) + SH) & (IMG_W - 1);
      size_t g = (((size_t)bimg * IMG_H + gh) * IMG_W + gw) * DIM + col;
      out[g] = fmap[g] + acc[r] + bval;
    }
  }
}

// ---------------------------------------------------------------------------
// Kernel 2: x += W2^T(gelu(W1^T(LN2(x)))), in place on d_out; 64 tokens/block
// ---------------------------------------------------------------------------
__global__ __launch_bounds__(256) void mlp_kernel(
    const _Float16* __restrict__ w1T, const _Float16* __restrict__ w2T,
    const float* __restrict__ b1, const float* __restrict__ b2,
    const float* __restrict__ g2, const float* __restrict__ be2,
    float* __restrict__ xio) {
  extern __shared__ __align__(16) char smem[];
  _Float16* xln = (_Float16*)smem;        // 64x192 f16
  _Float16* h1  = xln + 64 * DIM;         // 64x768 f16

  const size_t base = (size_t)blockIdx.x * WINL * DIM;
  const int tid  = threadIdx.x;
  const int lane = tid & 31;
  const int wave = tid >> 5;
  const int n16  = lane & 15;
  const int hb   = (lane >> 4) * 8;

  // ---- LayerNorm2 ---------------------------------------------------------
  {
    int tok = tid >> 2, q4 = tid & 3;
    const float* src = xio + base + (size_t)tok * DIM + q4 * 48;
    float4 v[12];
    float s = 0.f, ss = 0.f;
#pragma unroll
    for (int r = 0; r < 12; ++r) {
      v[r] = ((const float4*)src)[r];
      s  += v[r].x + v[r].y + v[r].z + v[r].w;
      ss += v[r].x * v[r].x + v[r].y * v[r].y + v[r].z * v[r].z +
            v[r].w * v[r].w;
    }
#pragma unroll
    for (int off = 1; off < 4; off <<= 1) {
      s  += __shfl_xor(s, off, 32);
      ss += __shfl_xor(ss, off, 32);
    }
    float mu  = s * (1.f / DIM);
    float inv = rsqrtf(ss * (1.f / DIM) - mu * mu + 1e-5f);
    _Float16* dst = xln + tok * DIM + q4 * 48;
#pragma unroll
    for (int r = 0; r < 12; ++r) {
      int c = q4 * 48 + r * 4;
      dst[r * 4 + 0] = (_Float16)((v[r].x - mu) * inv * g2[c + 0] + be2[c + 0]);
      dst[r * 4 + 1] = (_Float16)((v[r].y - mu) * inv * g2[c + 1] + be2[c + 1]);
      dst[r * 4 + 2] = (_Float16)((v[r].z - mu) * inv * g2[c + 2] + be2[c + 2]);
      dst[r * 4 + 3] = (_Float16)((v[r].w - mu) * inv * g2[c + 3] + be2[c + 3]);
    }
  }
  __syncthreads();

  // ---- H = gelu(X @ W1^T + b1): 48 column tasks of 16 --------------------
  for (int nt = wave; nt < 48; nt += 8) {
    v8f acc[4] = {};
#pragma unroll
    for (int ks = 0; ks < 6; ++ks) {
      v16h b = load_bt_frag(w1T, DIM, nt * 16, ks * 32, lane);
#pragma unroll
      for (int m = 0; m < 4; ++m)
        acc[m] = wmma32(load_a_frag(xln, DIM, m * 16, ks * 32, lane), b, acc[m]);
    }
    float bb = b1[nt * 16 + n16];
#pragma unroll
    for (int m = 0; m < 4; ++m)
#pragma unroll
      for (int r = 0; r < 8; ++r) {
        float v = acc[m][r] + bb;
        v = 0.5f * v * (1.f + erff(v * 0.70710678f));  // exact gelu
        h1[(m * 16 + hb + r) * HID + nt * 16 + n16] = (_Float16)v;
      }
  }
  __syncthreads();

  // ---- x += H @ W2^T + b2: 48 tiles, K = 768 ------------------------------
  for (int task = wave; task < 48; task += 8) {
    int mt = task / 12, nt = task - mt * 12;
    v8f acc = {};
#pragma unroll
    for (int ks = 0; ks < 24; ++ks) {
      __builtin_prefetch(w2T + (size_t)(nt * 16 + n16) * HID + ks * 32 + 64, 0, 1);
      v16h a = load_a_frag(h1, HID, mt * 16, ks * 32, lane);
      v16h b = load_bt_frag(w2T, HID, nt * 16, ks * 32, lane);
      acc = wmma32(a, b, acc);
    }
    int col = nt * 16 + n16;
    float bb = b2[col];
#pragma unroll
    for (int r = 0; r < 8; ++r) {
      size_t g = base + (size_t)(mt * 16 + hb + r) * DIM + col;
      xio[g] = xio[g] + acc[r] + bb;
    }
  }
}

// ---------------------------------------------------------------------------
// Host launcher
// ---------------------------------------------------------------------------
extern "C" void kernel_launch(void* const* d_in, const int* in_sizes, int n_in,
                              void* d_out, int out_size, void* d_ws,
                              size_t ws_size, hipStream_t stream) {
  (void)in_sizes; (void)n_in; (void)out_size; (void)ws_size;
  const float* fmap = (const float*)d_in[0];
  const float* wq = (const float*)d_in[1];
  const float* bq = (const float*)d_in[2];
  const float* wk = (const float*)d_in[3];
  const float* bk = (const float*)d_in[4];
  const float* wv = (const float*)d_in[5];
  const float* bv = (const float*)d_in[6];
  const float* wp = (const float*)d_in[7];
  const float* bp = (const float*)d_in[8];
  const float* w1 = (const float*)d_in[9];
  const float* b1 = (const float*)d_in[10];
  const float* w2 = (const float*)d_in[11];
  const float* b2 = (const float*)d_in[12];
  const float* g1 = (const float*)d_in[13];
  const float* be1 = (const float*)d_in[14];
  const float* g2 = (const float*)d_in[15];
  const float* be2 = (const float*)d_in[16];

  // workspace carve: transposed f16 weights (all 32B aligned)
  _Float16* wsp = (_Float16*)d_ws;
  _Float16* wqT = wsp;                 // [192][192]
  _Float16* wkT = wqT + DIM * DIM;
  _Float16* wvT = wkT + DIM * DIM;
  _Float16* wpT = wvT + DIM * DIM;
  _Float16* w1T = wpT + DIM * DIM;     // [768][192]
  _Float16* w2T = w1T + DIM * HID;     // [192][768]

  transpose_cvt<<<(DIM * DIM + 255) / 256, 256, 0, stream>>>(wq, wqT, DIM, DIM);
  transpose_cvt<<<(DIM * DIM + 255) / 256, 256, 0, stream>>>(wk, wkT, DIM, DIM);
  transpose_cvt<<<(DIM * DIM + 255) / 256, 256, 0, stream>>>(wv, wvT, DIM, DIM);
  transpose_cvt<<<(DIM * DIM + 255) / 256, 256, 0, stream>>>(wp, wpT, DIM, DIM);
  transpose_cvt<<<(DIM * HID + 255) / 256, 256, 0, stream>>>(w1, w1T, DIM, HID);
  transpose_cvt<<<(DIM * HID + 255) / 256, 256, 0, stream>>>(w2, w2T, HID, DIM);

  (void)hipFuncSetAttribute((const void*)attn_kernel,
                            hipFuncAttributeMaxDynamicSharedMemorySize, 122880);
  (void)hipFuncSetAttribute((const void*)mlp_kernel,
                            hipFuncAttributeMaxDynamicSharedMemorySize, 122880);

  const int nBlocksAttn = 2 * 32 * 32;  // B * nWy * nWx
  attn_kernel<<<nBlocksAttn, 256, 122880, stream>>>(
      fmap, wqT, wkT, wvT, wpT, bq, bk, bv, bp, g1, be1, (float*)d_out);

  const int nBlocksMlp = (2 * IMG_H * IMG_W) / WINL;  // 2048
  mlp_kernel<<<nBlocksMlp, 256, 122880, stream>>>(w1T, w2T, b1, b2, g2, be2,
                                                  (float*)d_out);
}
